// Attention_21440476741938
// MI455X (gfx1250) — compile-verified, matrix operand order checked by reading, not run on hardware
//
#include <hip/hip_runtime.h>
#include <stdint.h>

#define NHEADS 24
#define HDIM   64
#define BATCH  2
#define NIMG   4096
#define NTXT   128
#define CMOD   1536
#define CCTX   4096
#define ATT_SCALE 0.125f
#define NEG_BIG  -3.0e38f

typedef __attribute__((ext_vector_type(16))) __bf16 v16bf;
typedef __attribute__((ext_vector_type(8)))  float  v8f;
typedef unsigned int u32x4 __attribute__((ext_vector_type(4)));
typedef int          i32x4 __attribute__((ext_vector_type(4)));
typedef int          i32x8 __attribute__((ext_vector_type(8)));

union F8   { v8f v; float f[8]; };
union U4BF { uint4 u[2]; v16bf v; };
union U1H  { uint4 u; unsigned short h[8]; };

__device__ __forceinline__ unsigned short f2bf(float x) {
  unsigned int u = __float_as_uint(x);
  u += 0x7FFFu + ((u >> 16) & 1u);   // round-to-nearest-even
  return (unsigned short)(u >> 16);
}

// Fragment loader: two contiguous 16B groups, 16 bf16 apart (WMMA bf16 A/B
// per-lane layout: k = (e<8?0:16) + khalf*8 + (e&7); caller bakes khalf*8 into p).
__device__ __forceinline__ v16bf ld_frag(const unsigned short* p) {
  U4BF r;
  r.u[0] = *(const uint4*)(p);
  r.u[1] = *(const uint4*)(p + 16);
  return r.v;
}

// ---------------- fp32 -> bf16 conversion (8 elems / thread) ----------------
__global__ void cvt_kernel(const float* __restrict__ in,
                           unsigned short* __restrict__ out, int n8) {
  int i = blockIdx.x * blockDim.x + threadIdx.x;
  if (i >= n8) return;
  const float4* p = (const float4*)in + (size_t)i * 2;
  float4 a = p[0], b = p[1];
  U1H o;
  o.h[0] = f2bf(a.x); o.h[1] = f2bf(a.y); o.h[2] = f2bf(a.z); o.h[3] = f2bf(a.w);
  o.h[4] = f2bf(b.x); o.h[5] = f2bf(b.y); o.h[6] = f2bf(b.z); o.h[7] = f2bf(b.w);
  ((uint4*)out)[i] = o.u;
}

// ---------------- bf16 WMMA GEMM: C(MxN) = A(MxK) @ B(KxN) ----------------
// 128x128 block tile, 256 threads = 8 waves (4 M x 2 N), wave tile 32x64.
// K-step 64, double-buffered LDS: next-stage global loads issued before the
// 16 WMMAs of the current stage, drained to the alternate buffer after.
// Requires M%128==0, N%128==0, K%64==0 (true for all uses here).
template<bool OUT_BF16>
__global__ __launch_bounds__(256) void gemm_bf16_kernel(
    const unsigned short* __restrict__ A, const unsigned short* __restrict__ Bm,
    void* __restrict__ C, int M, int N, int K) {
  __shared__ unsigned short Asl[2][128 * 72];   // [m][k], stride 72 bf16 (144B)
  __shared__ unsigned short Bsl[2][128 * 72];   // [n][k] transposed, stride 72

  const int tid  = threadIdx.x;
  const int lane = tid & 31, wid = tid >> 5;
  const int wm = wid & 3, wn = wid >> 2;
  const int m0 = blockIdx.y * 128, n0 = blockIdx.x * 128;
  const int l15 = lane & 15, kh = lane >> 4;

  F8 acc[2][4];
  #pragma unroll
  for (int sm = 0; sm < 2; ++sm)
    #pragma unroll
    for (int sn = 0; sn < 4; ++sn)
      #pragma unroll
      for (int j = 0; j < 8; ++j) acc[sm][sn].f[j] = 0.0f;

  const int nk = K >> 6;

  // prologue: stage 0 straight into buffer 0
  #pragma unroll
  for (int i = 0; i < 4; ++i) {
    int c = tid + i * 256;                  // 1024 chunks of 8 bf16 (A: 128x64)
    int row = c >> 3, kc = c & 7;
    uint4 d = *(const uint4*)(A + (size_t)(m0 + row) * K + kc * 8);
    *(uint4*)(&Asl[0][row * 72 + kc * 8]) = d;
  }
  #pragma unroll
  for (int i = 0; i < 4; ++i) {
    int c = tid + i * 256;                  // B: 64x128, transpose into [n][k]
    int kr = c >> 4, nc = c & 15;
    U1H d; d.u = *(const uint4*)(Bm + (size_t)kr * N + n0 + nc * 8);
    #pragma unroll
    for (int e = 0; e < 8; ++e) Bsl[0][(nc * 8 + e) * 72 + kr] = d.h[e];
  }
  __syncthreads();

  for (int kt = 0; kt < nk; ++kt) {
    const int cur = kt & 1;
    const bool more = (kt + 1 < nk);
    uint4 ar[4]; U1H br[4];
    if (more) {
      const int k0n = (kt + 1) << 6;
      #pragma unroll
      for (int i = 0; i < 4; ++i) {
        int c = tid + i * 256;
        int row = c >> 3, kc = c & 7;
        ar[i] = *(const uint4*)(A + (size_t)(m0 + row) * K + k0n + kc * 8);
      }
      #pragma unroll
      for (int i = 0; i < 4; ++i) {
        int c = tid + i * 256;
        int kr = c >> 4, nc = c & 15;
        br[i].u = *(const uint4*)(Bm + (size_t)(k0n + kr) * N + n0 + nc * 8);
      }
      if (kt + 2 < nk) {      // pre-warm L2 for the stage after next
        const int k0p = (kt + 2) << 6;
        __builtin_prefetch(A + (size_t)(m0 + (tid >> 1)) * K + k0p + (tid & 1) * 32, 0, 1);
        __builtin_prefetch(Bm + (size_t)(k0p + (tid >> 2)) * N + n0 + (tid & 3) * 32, 0, 1);
      }
    }
    // 16 WMMAs on the current buffer
    #pragma unroll
    for (int ks = 0; ks < 2; ++ks) {
      v16bf af[2], bfr[4];
      #pragma unroll
      for (int sm = 0; sm < 2; ++sm)
        af[sm] = ld_frag(&Asl[cur][(wm * 32 + sm * 16 + l15) * 72 + ks * 32 + kh * 8]);
      #pragma unroll
      for (int sn = 0; sn < 4; ++sn)
        bfr[sn] = ld_frag(&Bsl[cur][(wn * 64 + sn * 16 + l15) * 72 + ks * 32 + kh * 8]);
      #pragma unroll
      for (int sm = 0; sm < 2; ++sm)
        #pragma unroll
        for (int sn = 0; sn < 4; ++sn)
          acc[sm][sn].v = __builtin_amdgcn_wmma_f32_16x16x32_bf16(
              false, af[sm], false, bfr[sn], (short)0, acc[sm][sn].v, false, false);
    }
    // drain staged registers into the alternate buffer
    if (more) {
      const int nxt = cur ^ 1;
      #pragma unroll
      for (int i = 0; i < 4; ++i) {
        int c = tid + i * 256;
        int row = c >> 3, kc = c & 7;
        *(uint4*)(&Asl[nxt][row * 72 + kc * 8]) = ar[i];
      }
      #pragma unroll
      for (int i = 0; i < 4; ++i) {
        int c = tid + i * 256;
        int kr = c >> 4, nc = c & 15;
        #pragma unroll
        for (int e = 0; e < 8; ++e) Bsl[nxt][(nc * 8 + e) * 72 + kr] = br[i].h[e];
      }
    }
    __syncthreads();
  }

  const int mloc = (lane < 16) ? 0 : 8;    // C layout: m = j (+8 for hi lanes)
  #pragma unroll
  for (int sm = 0; sm < 2; ++sm)
    #pragma unroll
    for (int sn = 0; sn < 4; ++sn)
      #pragma unroll
      for (int j = 0; j < 8; ++j) {
        size_t m = m0 + wm * 32 + sm * 16 + mloc + j;
        size_t n = n0 + wn * 64 + sn * 16 + l15;
        if (OUT_BF16) ((unsigned short*)C)[m * N + n] = f2bf(acc[sm][sn].f[j]);
        else          ((float*)C)[m * N + n] = acc[sm][sn].f[j];
      }
}

// ---------------- fused attention: per (b,head), 128 q rows / block ----------
// K tile staged by the Tensor Data Mover (2D tile + LDS row padding),
// V staged transposed manually. Each wave: 16 q rows, 16 WMMA scores,
// masked softmax via 16-lane shfl reductions, 16 WMMA for P@V.
__global__ __launch_bounds__(256) void attn_kernel(
    const unsigned short* __restrict__ q,   // (B, NIMG, 1536) bf16
    const unsigned short* __restrict__ k,   // (B, NTXT, 1536) bf16
    const unsigned short* __restrict__ v,   // (B, NTXT, 1536) bf16
    const unsigned char*  __restrict__ mask,// (B, NTXT)
    unsigned short* __restrict__ o)         // (B, NIMG, 1536) bf16
{
  __shared__ unsigned short Ksl[NTXT * 72];        // [tok][d],  stride 72 (144B)
  __shared__ unsigned short VTsl[HDIM * 136];      // [d][tok],  stride 136
  __shared__ unsigned short Psl[8 * 16 * 136];     // per-wave P[m][tok]

  const int bh = blockIdx.x;
  const int b = bh / NHEADS, h = bh % NHEADS;
  const int qbase = blockIdx.y * 128;
  const int tid = threadIdx.x, lane = tid & 31, wid = tid >> 5;
  const int l15 = lane & 15, kh = lane >> 4;

  // ---- TDM: DMA the K tile (128 tok x 64 d bf16, global row stride 1536) ----
  // pad_interval = 32 DWORDs (one 128B row), pad_amount = 4 DWORDs (16B)
  // => LDS row stride 144B = 72 bf16, matching Ksl.
  if (wid == 0) {
    const unsigned lds_off = (unsigned)(unsigned long long)(const void*)&Ksl[0];
    const unsigned long long ga =
        (unsigned long long)(const void*)(k + (size_t)(b * NTXT) * CMOD + h * HDIM);
    u32x4 g0;
    g0[0] = 1u;                                            // count=1 (valid D#)
    g0[1] = lds_off;                                       // lds_addr
    g0[2] = (unsigned)(ga & 0xFFFFFFFFu);                  // global_addr[31:0]
    g0[3] = (unsigned)((ga >> 32) & 0x01FFFFFFu) | (2u << 30); // [56:32]|type=2
    i32x8 g1;
    g1[0] = (int)((1u << 16) | (1u << 20) | (4u << 22) | (3u << 25));
            // data_size=2B | pad_enable | pad_interval=32dw | pad_amount=4dw
    g1[1] = (int)(((unsigned)CMOD & 0xFFFFu) << 16);       // tensor_dim0 lo16
    g1[2] = (int)((((unsigned)CMOD >> 16) & 0xFFFFu) |
                  (((unsigned)(BATCH * NTXT) & 0xFFFFu) << 16)); // dim0 hi|dim1
    g1[3] = (int)((unsigned)HDIM << 16);                   // tile_dim0 = 64
    g1[4] = (int)(unsigned)NTXT;                           // tile_dim1=128, tile_dim2=0
    g1[5] = (int)(unsigned)CMOD;                           // tensor_dim0_stride
    g1[6] = 0;
    g1[7] = 0;
    i32x4 z4 = {0, 0, 0, 0};
#if defined(__clang_major__) && (__clang_major__ >= 23)
    i32x8 z8 = {0, 0, 0, 0, 0, 0, 0, 0};
    __builtin_amdgcn_tensor_load_to_lds(g0, g1, z4, z4, z8, 0);
#else
    __builtin_amdgcn_tensor_load_to_lds(g0, g1, z4, z4, 0);
#endif
  }

  // stage V transposed [d][tok] (TDM cannot transpose)
  #pragma unroll
  for (int i = 0; i < 32; ++i) {
    int e = tid + i * 256;                 // 8192 elements
    int tok = e >> 6, d = e & 63;
    VTsl[d * 136 + tok] = v[(size_t)(b * NTXT + tok) * CMOD + h * HDIM + d];
  }

  if (wid == 0) __builtin_amdgcn_s_wait_tensorcnt(0);
  __syncthreads();

  // Q fragments straight from global (2 K-steps over d=64)
  const int qrow = qbase + wid * 16 + l15;
  const size_t qoff = ((size_t)b * NIMG + qrow) * CMOD + h * HDIM;
  v16bf aq[2];
  aq[0] = ld_frag(q + qoff + 0  + kh * 8);
  aq[1] = ld_frag(q + qoff + 32 + kh * 8);

  bool valid[8];
  #pragma unroll
  for (int t = 0; t < 8; ++t) valid[t] = mask[b * NTXT + t * 16 + l15] != 0;

  // scores: 8 column tiles of 16 tokens
  F8 sc[8];
  #pragma unroll
  for (int t = 0; t < 8; ++t)
    #pragma unroll
    for (int j = 0; j < 8; ++j) sc[t].f[j] = 0.0f;
  #pragma unroll
  for (int t = 0; t < 8; ++t)
    #pragma unroll
    for (int ks = 0; ks < 2; ++ks) {
      v16bf bk = ld_frag(&Ksl[(t * 16 + l15) * 72 + ks * 32 + kh * 8]);
      sc[t].v = __builtin_amdgcn_wmma_f32_16x16x32_bf16(
          false, aq[ks], false, bk, (short)0, sc[t].v, false, false);
    }

  // mask + scale + softmax; row m of C lives across 16 lanes of one VGPR
  const int mloc = (lane < 16) ? 0 : 8;
  #pragma unroll
  for (int j = 0; j < 8; ++j) {
    float mx = NEG_BIG;
    #pragma unroll
    for (int t = 0; t < 8; ++t) {
      float s = valid[t] ? sc[t].f[j] * ATT_SCALE : NEG_BIG;
      sc[t].f[j] = s;
      mx = fmaxf(mx, s);
    }
    #pragma unroll
    for (int off = 8; off > 0; off >>= 1) mx = fmaxf(mx, __shfl_xor(mx, off, 16));
    float sum = 0.0f;
    #pragma unroll
    for (int t = 0; t < 8; ++t) {
      float p = __expf(sc[t].f[j] - mx);
      sc[t].f[j] = p;
      sum += p;
    }
    #pragma unroll
    for (int off = 8; off > 0; off >>= 1) sum += __shfl_xor(sum, off, 16);
    float r = 1.0f / sum;
    int m = mloc + j;
    #pragma unroll
    for (int t = 0; t < 8; ++t)
      Psl[(wid * 16 + m) * 136 + t * 16 + l15] = f2bf(sc[t].f[j] * r);
  }
  // wave reads back only its own P region; dscnt ordering handled by compiler

  // out = P @ V : 4 K-steps over 128 tokens, 4 N-tiles over d=64
  F8 oacc[4];
  #pragma unroll
  for (int nt = 0; nt < 4; ++nt)
    #pragma unroll
    for (int j = 0; j < 8; ++j) oacc[nt].f[j] = 0.0f;
  #pragma unroll
  for (int ks = 0; ks < 4; ++ks) {
    v16bf ap = ld_frag(&Psl[(wid * 16 + l15) * 136 + ks * 32 + kh * 8]);
    #pragma unroll
    for (int nt = 0; nt < 4; ++nt) {
      v16bf bv = ld_frag(&VTsl[(nt * 16 + l15) * 136 + ks * 32 + kh * 8]);
      oacc[nt].v = __builtin_amdgcn_wmma_f32_16x16x32_bf16(
          false, ap, false, bv, (short)0, oacc[nt].v, false, false);
    }
  }
  #pragma unroll
  for (int nt = 0; nt < 4; ++nt)
    #pragma unroll
    for (int j = 0; j < 8; ++j) {
      size_t m = (size_t)qbase + wid * 16 + mloc + j;
      o[((size_t)b * NIMG + m) * CMOD + h * HDIM + nt * 16 + l15] =
          f2bf(oacc[nt].f[j]);
    }
}

// ---------------------------------------------------------------------------
extern "C" void kernel_launch(void* const* d_in, const int* in_sizes, int n_in,
                              void* d_out, int out_size, void* d_ws, size_t ws_size,
                              hipStream_t stream) {
  const float* x   = (const float*)d_in[0];
  const float* ctx = (const float*)d_in[1];
  const unsigned char* mask = (const unsigned char*)d_in[2];
  const float* Wq  = (const float*)d_in[3];
  const float* Wk  = (const float*)d_in[4];
  const float* Wv  = (const float*)d_in[5];
  const float* Wo  = (const float*)d_in[6];
  float* out = (float*)d_out;

  char* ws = (char*)d_ws;
  size_t off = 0;
  auto alloc = [&](size_t elems) {
    char* p = ws + off;
    off += (elems * sizeof(unsigned short) + 255) & ~(size_t)255;
    return (unsigned short*)p;
  };
  const size_t n_x  = (size_t)BATCH * NIMG * CMOD;   // 12.58M
  const size_t n_c  = (size_t)BATCH * NTXT * CCTX;   //  1.05M
  const size_t n_wq = (size_t)CMOD * CMOD;
  const size_t n_wk = (size_t)CCTX * CMOD;
  const size_t n_kv = (size_t)BATCH * NTXT * CMOD;

  unsigned short* xb  = alloc(n_x);
  unsigned short* cb  = alloc(n_c);
  unsigned short* wqb = alloc(n_wq);
  unsigned short* wkb = alloc(n_wk);
  unsigned short* wvb = alloc(n_wk);
  unsigned short* wob = alloc(n_wq);
  unsigned short* qb  = alloc(n_x);
  unsigned short* kb  = alloc(n_kv);
  unsigned short* vb  = alloc(n_kv);
  unsigned short* ob  = alloc(n_x);
  (void)ws_size; (void)in_sizes; (void)n_in; (void)out_size;

  auto cvt = [&](const float* src, unsigned short* dst, size_t n) {
    int n8 = (int)(n / 8);
    cvt_kernel<<<(n8 + 255) / 256, 256, 0, stream>>>(src, dst, n8);
  };
  cvt(x,   xb,  n_x);
  cvt(ctx, cb,  n_c);
  cvt(Wq,  wqb, n_wq);
  cvt(Wk,  wkb, n_wk);
  cvt(Wv,  wvb, n_wk);
  cvt(Wo,  wob, n_wq);

  // Q = x @ Wq   (8192 x 1536 x 1536)
  gemm_bf16_kernel<true><<<dim3(CMOD / 128, (BATCH * NIMG) / 128), 256, 0, stream>>>(
      xb, wqb, qb, BATCH * NIMG, CMOD, CMOD);
  // K = ctx @ Wk (256 x 1536 x 4096)
  gemm_bf16_kernel<true><<<dim3(CMOD / 128, (BATCH * NTXT) / 128), 256, 0, stream>>>(
      cb, wkb, kb, BATCH * NTXT, CMOD, CCTX);
  // V = ctx @ Wv
  gemm_bf16_kernel<true><<<dim3(CMOD / 128, (BATCH * NTXT) / 128), 256, 0, stream>>>(
      cb, wvb, vb, BATCH * NTXT, CMOD, CCTX);

  // fused attention: 48 (b,head) x 32 q-tiles of 128 rows
  attn_kernel<<<dim3(BATCH * NHEADS, NIMG / 128), 256, 0, stream>>>(
      qb, kb, vb, mask, ob);

  // out = attn_out @ Wo, fp32 result
  gemm_bf16_kernel<false><<<dim3(CMOD / 128, (BATCH * NIMG) / 128), 256, 0, stream>>>(
      ob, wob, out, BATCH * NIMG, CMOD, CMOD);
}
